// FullModel_79456894976422
// MI455X (gfx1250) — matrix operand order repaired and checked by reading, not compiled, not run on hardware
//
#include <hip/hip_runtime.h>
#include <hip/hip_bf16.h>

typedef __attribute__((ext_vector_type(16))) _Float16 v16h;
typedef __attribute__((ext_vector_type(8)))  _Float16 v8h;
typedef __attribute__((ext_vector_type(8)))  float    v8f;
typedef __attribute__((ext_vector_type(4)))  unsigned int u32x4;
typedef __attribute__((ext_vector_type(8)))  int      i32x8;
typedef __attribute__((ext_vector_type(4)))  int      i32x4;

#define BSZ 16384
#define NSTEPS 60
#define LDY 128
#define BN_EPS 1e-6f

// ---------------------------------------------------------------------------
// init: y <- broadcast(y_true), zero stat accumulators, zero scalar output
// ---------------------------------------------------------------------------
__global__ void init_kernel(float* __restrict__ y, const float* __restrict__ ytrue,
                            float* __restrict__ zeroArea, int zeroN, float* __restrict__ out) {
    int idx = blockIdx.x * blockDim.x + threadIdx.x;
    if (idx < BSZ * LDY) {
        int c = idx & (LDY - 1);
        y[idx] = (c < 120) ? ytrue[c] : 0.0f;
    }
    if (idx < zeroN) zeroArea[idx] = 0.0f;
    if (idx == 0) *out = 0.0f;
}

// ---------------------------------------------------------------------------
// column stats of v (61 cols) and x (61 cols): sums -> [vs|vss|xs|xss] (64 ea)
// ---------------------------------------------------------------------------
__global__ void vxstats_acc(const float* __restrict__ v, const float* __restrict__ x,
                            float* __restrict__ sums) {
    int c = threadIdx.x & 63;
    int q = threadIdx.x >> 6;
    if (c >= 61) return;
    int b0 = blockIdx.x * 256 + q * 64;
    float s1 = 0.f, s2 = 0.f, s3 = 0.f, s4 = 0.f;
    for (int r = 0; r < 64; ++r) {
        int b = b0 + r;
        float vv = v[b * 61 + c]; s1 += vv; s2 += vv * vv;
        float xv = x[b * 61 + c]; s3 += xv; s4 += xv * xv;
    }
    atomicAdd(&sums[c],       s1);
    atomicAdd(&sums[64 + c],  s2);
    atomicAdd(&sums[128 + c], s3);
    atomicAdd(&sums[192 + c], s4);
}

__global__ void vxstats_fin(const float* __restrict__ sums, float* __restrict__ st) {
    int c = threadIdx.x;
    if (c >= 128) return;
    int idx = (c < 64) ? c : (c + 64);
    float s1 = sums[idx], s2 = sums[idx + 64];
    float mean = s1 * (1.0f / (float)BSZ);
    float var  = s2 * (1.0f / (float)BSZ) - mean * mean;
    st[idx]      = mean;
    st[idx + 64] = rsqrtf(fmaxf(var, 0.0f) + BN_EPS);
}

// ---------------------------------------------------------------------------
// Layer 1: BN0(concat(v[:,:i+1], x[:,i])) @ W0 + b0 -> H1 (B x 32), BN1 stats.
// LDS-staged f16 A (BN0 folded during staging) + transposed W; each wave
// computes two 16x16 tiles (n and n+16) reusing its A fragment.
// ---------------------------------------------------------------------------
__global__ void gemm1_kernel(const float* __restrict__ v, const float* __restrict__ x,
                             const float* __restrict__ vxm,
                             const float* __restrict__ g0, const float* __restrict__ be0,
                             const float* __restrict__ W0, const float* __restrict__ b0,
                             float* __restrict__ H1, float* __restrict__ stats,
                             int step, int in_dim) {
    __shared__ __align__(16) _Float16 sA[128 * 72];   // rows x Kpad(64), stride 72
    __shared__ __align__(16) _Float16 sW[32 * 72];    // [n][k] transposed, stride 72
    __shared__ float sAff[128];                       // (scale, shift) per K column

    int t = threadIdx.x;
    if (t < 64) {
        float sc = 0.f, sh = 0.f;
        if (t < in_dim) {
            float m, rs;
            if (t <= step) { m = vxm[t];          rs = vxm[64 + t]; }
            else           { m = vxm[128 + step]; rs = vxm[192 + step]; }
            sc = g0[t] * rs;
            sh = be0[t] - sc * m;
        }
        sAff[2 * t] = sc; sAff[2 * t + 1] = sh;
    }
    __syncthreads();

    int rowBase = blockIdx.x * 128;
    // stage transformed input (coalesced over k)
    for (int idx = t; idx < 128 * 64; idx += 256) {
        int r = idx >> 6, k = idx & 63;
        float raw = 0.0f;
        if (k <= step)          raw = v[(rowBase + r) * 61 + k];
        else if (k == step + 1) raw = x[(rowBase + r) * 61 + step];
        sA[r * 72 + k] = (_Float16)(sAff[2 * k] * raw + sAff[2 * k + 1]);
    }
    // stage weights transposed, zero-padded in K
    for (int idx = t; idx < 32 * 64; idx += 256) {
        int k = idx >> 5, n = idx & 31;
        float wv = (k < in_dim) ? W0[k * 32 + n] : 0.0f;
        sW[n * 72 + k] = (_Float16)wv;
    }
    __syncthreads();

    int lane = t & 31, wave = t >> 5;
    int half = lane >> 4, Nl = lane & 15;
    int mloc = wave * 16 + Nl;

    v8f c0 = {}, c1 = {};
    for (int kk = 0; kk < 64; kk += 32) {
        if (kk >= in_dim) break;                      // wave-uniform
        v8h alo = *(const v8h*)&sA[mloc * 72 + kk + half * 8];
        v8h ahi = *(const v8h*)&sA[mloc * 72 + kk + 16 + half * 8];
        v16h a;
        #pragma unroll
        for (int e = 0; e < 8; ++e) { a[e] = alo[e]; a[e + 8] = ahi[e]; }

        v8h b0lo = *(const v8h*)&sW[Nl * 72 + kk + half * 8];
        v8h b0hi = *(const v8h*)&sW[Nl * 72 + kk + 16 + half * 8];
        v8h b1lo = *(const v8h*)&sW[(16 + Nl) * 72 + kk + half * 8];
        v8h b1hi = *(const v8h*)&sW[(16 + Nl) * 72 + kk + 16 + half * 8];
        v16h bm0, bm1;
        #pragma unroll
        for (int e = 0; e < 8; ++e) {
            bm0[e] = b0lo[e]; bm0[e + 8] = b0hi[e];
            bm1[e] = b1lo[e]; bm1[e + 8] = b1hi[e];
        }
        c0 = __builtin_amdgcn_wmma_f32_16x16x32_f16(false, a, false, bm0, (short)0, c0, false, false);
        c1 = __builtin_amdgcn_wmma_f32_16x16x32_f16(false, a, false, bm1, (short)0, c1, false, false);
    }

    int rtop = rowBase + wave * 16 + half * 8;
    float bia0 = b0[Nl], bia1 = b0[16 + Nl];
    float s1a = 0.f, s2a = 0.f, s1b = 0.f, s2b = 0.f;
    #pragma unroll
    for (int r = 0; r < 8; ++r) {
        float y0 = c0[r] + bia0;
        float y1 = c1[r] + bia1;
        H1[(rtop + r) * 32 + Nl]      = y0;
        H1[(rtop + r) * 32 + 16 + Nl] = y1;
        s1a += y0; s2a += y0 * y0;
        s1b += y1; s2b += y1 * y1;
    }
    s1a += __shfl_xor(s1a, 16, 32); s2a += __shfl_xor(s2a, 16, 32);
    s1b += __shfl_xor(s1b, 16, 32); s2b += __shfl_xor(s2b, 16, 32);
    if (half == 0) {
        atomicAdd(&stats[Nl],            s1a);
        atomicAdd(&stats[128 + Nl],      s2a);
        atomicAdd(&stats[16 + Nl],       s1b);
        atomicAdd(&stats[128 + 16 + Nl], s2b);
    }
}

// ---------------------------------------------------------------------------
// Layers 2/3: Y = leaky(aff(X)) @ W + b ; X is B x 32 f32, TDM-staged to LDS
// with padding to a 36-float row stride. Each wave computes all Cp/16 column
// tiles reusing one A fragment. Accumulates next-BN column stats.
// ---------------------------------------------------------------------------
__global__ void gemm32_kernel(const float* __restrict__ X, const float* __restrict__ aff,
                              const float* __restrict__ W, const float* __restrict__ bias,
                              float* __restrict__ Y, float* __restrict__ stats,
                              int Cout, int Cp) {
    __shared__ __align__(16) float    sX[128 * 36];   // 128 rows x 32 f32, stride 36
    __shared__ __align__(16) _Float16 sW[128 * 64];   // [n][k] transposed, stride 64
    __shared__ float sAff[64];
    __shared__ float sBias[128];

    int t = threadIdx.x;
    int rowBase = blockIdx.x * 128;

    // --- Tensor Data Mover: X[rowBase .. rowBase+127][0..31] -> sX (padded) ---
    if ((t >> 5) == 0) {
        unsigned long long gaddr = (unsigned long long)(X + (size_t)rowBase * 32);
        unsigned int ldsOff = (unsigned int)(size_t)&sX[0];
        u32x4 g0;
        g0[0] = 1u;                                   // count=1, is_restore=0
        g0[1] = ldsOff;                               // lds_addr
        g0[2] = (unsigned int)(gaddr & 0xFFFFFFFFu);  // global_addr lo
        g0[3] = (unsigned int)((gaddr >> 32) & 0x1FFFFFFu) | (2u << 30); // type=2
        i32x8 g1;
        // data_size=4B(2), pad_enable, pad_interval=32DW(code4), pad_amount=4DW(code3)
        g1[0] = (2 << 16) | (1 << 20) | (4 << 22) | (3 << 25);
        g1[1] = (32 & 0xFFFF) << 16;                  // tensor_dim0 = 32 (bits 79:48 lo)
        g1[2] = ((128 & 0xFFFF) << 16);               // tensor_dim1 = 128 (bits 111:80 lo)
        g1[3] = (32 << 16);                           // tile_dim0 = 32
        g1[4] = 128;                                  // tile_dim1 = 128, tile_dim2 = 0
        g1[5] = 32;                                   // tensor_dim0_stride = 32
        g1[6] = 0;
        g1[7] = 0;
        i32x4 gz4 = {0, 0, 0, 0};
        i32x8 gz8 = {0, 0, 0, 0, 0, 0, 0, 0};
        __builtin_amdgcn_tensor_load_to_lds(g0, g1, gz4, gz4, gz8, 0);
        __builtin_amdgcn_s_wait_tensorcnt((short)0);
    }
    // stage affine / bias / transposed weights (other waves work in parallel)
    if (t < 64)  sAff[t]  = aff[t];
    if (t < 128) sBias[t] = (t < Cout) ? bias[t] : 0.0f;
    for (int base = 0; base < Cp; base += 8) {
        int k = t >> 3;                               // 0..31
        int n = base + (t & 7);
        float wv = (n < Cout) ? W[k * Cout + n] : 0.0f;
        sW[n * 64 + k] = (_Float16)wv;
    }
    __syncthreads();

    int lane = t & 31, wave = t >> 5;
    int half = lane >> 4, Nl = lane & 15;
    int mloc = wave * 16 + Nl;

    // A fragment: 4 aligned 16B LDS reads + fused BN-affine + leaky + cvt f16
    const float* rowp = &sX[mloc * 36];
    float4 x0 = *(const float4*)(rowp + half * 8);
    float4 x1 = *(const float4*)(rowp + half * 8 + 4);
    float4 x2 = *(const float4*)(rowp + 16 + half * 8);
    float4 x3 = *(const float4*)(rowp + 16 + half * 8 + 4);
    float xs[16] = {x0.x, x0.y, x0.z, x0.w, x1.x, x1.y, x1.z, x1.w,
                    x2.x, x2.y, x2.z, x2.w, x3.x, x3.y, x3.z, x3.w};
    v16h a;
    #pragma unroll
    for (int e = 0; e < 16; ++e) {
        int ka = (e < 8) ? half * 8 + e : 16 + half * 8 + (e - 8);
        float av = sAff[2 * ka] * xs[e] + sAff[2 * ka + 1];
        av = (av < 0.0f) ? 0.3f * av : av;            // leaky_relu(0.3)
        a[e] = (_Float16)av;
    }

    int rtop = rowBase + wave * 16 + half * 8;
    for (int nt = 0; nt < (Cp >> 4); ++nt) {
        int n = nt * 16 + Nl;
        v8h blo = *(const v8h*)&sW[n * 64 + half * 8];
        v8h bhi = *(const v8h*)&sW[n * 64 + 16 + half * 8];
        v16h bm;
        #pragma unroll
        for (int e = 0; e < 8; ++e) { bm[e] = blo[e]; bm[e + 8] = bhi[e]; }
        v8f c = {};
        c = __builtin_amdgcn_wmma_f32_16x16x32_f16(false, a, false, bm, (short)0, c, false, false);

        float bia = sBias[n];
        float s1 = 0.f, s2 = 0.f;
        #pragma unroll
        for (int r = 0; r < 8; ++r) {
            float yv = c[r] + bia;
            Y[(rtop + r) * Cp + n] = yv;
            s1 += yv; s2 += yv * yv;
        }
        s1 += __shfl_xor(s1, 16, 32);
        s2 += __shfl_xor(s2, 16, 32);
        if (half == 0) {
            atomicAdd(&stats[n],       s1);
            atomicAdd(&stats[128 + n], s2);
        }
    }
}

// ---------------------------------------------------------------------------
// sums -> fused BN affine (scale, shift); zeroes stats for reuse
// ---------------------------------------------------------------------------
__global__ void finalize_bn(float* __restrict__ stats, const float* __restrict__ gamma,
                            const float* __restrict__ beta, float* __restrict__ aff, int C) {
    int c = threadIdx.x;
    if (c >= 128) return;
    float sc = 0.f, sh = 0.f;
    if (c < C) {
        float mean = stats[c] * (1.0f / (float)BSZ);
        float var  = stats[128 + c] * (1.0f / (float)BSZ) - mean * mean;
        sc = gamma[c] * rsqrtf(fmaxf(var, 0.0f) + BN_EPS);
        sh = beta[c] - sc * mean;
    }
    aff[2 * c] = sc; aff[2 * c + 1] = sh;
    stats[c] = 0.0f; stats[128 + c] = 0.0f;
}

// ---------------------------------------------------------------------------
// y[:, off+c] = (1+R*TAU)*y + BN3(tZ)*w_i + BN3(Z)*w_perp_i
// ---------------------------------------------------------------------------
__global__ void y_update(float* __restrict__ y,
                         const float* __restrict__ hz, const float* __restrict__ ht,
                         const float* __restrict__ affZ, const float* __restrict__ affT,
                         const float* __restrict__ w, const float* __restrict__ wp,
                         int step, int off, int ldh) {
    int b = blockIdx.x * blockDim.x + threadIdx.x;
    int c = blockIdx.y;
    float z  = affZ[2 * c] * hz[b * ldh + c] + affZ[2 * c + 1];
    float tz = affT[2 * c] * ht[b * ldh + c] + affT[2 * c + 1];
    float* yp = &y[b * LDY + off + c];
    const float growth = 1.0f + 0.05f / 60.0f;
    *yp = growth * (*yp) + tz * w[b * 60 + step] + z * wp[b * 60 + step];
}

// ---------------------------------------------------------------------------
// per-block loss
// ---------------------------------------------------------------------------
__global__ void loss_kernel(const float* __restrict__ x, const float* __restrict__ Kv,
                            const float* __restrict__ y, float* __restrict__ out, int j) {
    __shared__ float red[256];
    float acc = 0.0f;
    float rT = 0.05f * (float)j / 12.0f;
    int off = (j - 1) * 10;
    for (int b = blockIdx.x * blockDim.x + threadIdx.x; b < BSZ;
         b += gridDim.x * blockDim.x) {
        float s = expf(x[b * 61 + j * 5] + rT);
        for (int l = 0; l < 10; ++l) {
            float p = fmaxf(s - Kv[l], 0.0f);
            float d = p - y[b * LDY + off + l];
            acc += d * d;
        }
    }
    red[threadIdx.x] = acc;
    __syncthreads();
    for (int s = 128; s > 0; s >>= 1) {
        if (threadIdx.x < (unsigned)s) red[threadIdx.x] += red[threadIdx.x + s];
        __syncthreads();
    }
    if (threadIdx.x == 0)
        atomicAdd(out, red[0] * (1.0f / (12.0f * 10.0f * (float)BSZ)));
}

// ---------------------------------------------------------------------------
extern "C" void kernel_launch(void* const* d_in, const int* in_sizes, int n_in,
                              void* d_out, int out_size, void* d_ws, size_t ws_size,
                              hipStream_t stream) {
    (void)in_sizes; (void)n_in; (void)out_size; (void)ws_size;
    const float* w  = (const float*)d_in[0];
    const float* wp = (const float*)d_in[1];
    const float* v  = (const float*)d_in[2];
    const float* x  = (const float*)d_in[3];
    const float* yt = (const float*)d_in[4];
    const float* Kv = (const float*)d_in[5];
    auto leaf = [&](int sub, int i, int which) -> const float* {
        return (const float*)d_in[6 + sub * NSTEPS * 14 + i * 14 + which];
    };

    float* ws    = (float*)d_ws;
    float* y     = ws;
    float* hz1   = y   + (size_t)BSZ * LDY;
    float* ht1   = hz1 + (size_t)BSZ * 32;
    float* hz2   = ht1 + (size_t)BSZ * 32;
    float* ht2   = hz2 + (size_t)BSZ * 32;
    float* hz3   = ht2 + (size_t)BSZ * 32;
    float* ht3   = hz3 + (size_t)BSZ * LDY;
    float* stats = ht3 + (size_t)BSZ * LDY;      // 6 x 256
    float* aff   = stats + 6 * 256;              // 6 x 256
    float* vxs   = aff + 6 * 256;                // 256 sums
    float* vxm   = vxs + 256;                    // 256 mean/rstd
    float* out   = (float*)d_out;

    init_kernel<<<(BSZ * LDY) / 256, 256, 0, stream>>>(y, yt, stats, 6 * 256 + 256, out);
    vxstats_acc<<<64, 256, 0, stream>>>(v, x, vxs);
    vxstats_fin<<<1, 128, 0, stream>>>(vxs, vxm);

    for (int i = 0; i < NSTEPS; ++i) {
        int in_dim = i + 2;
        int j    = i / 5;
        int outd = (12 - j) * 10;
        int Cp   = (outd + 15) & ~15;
        int off  = j * 10;

        gemm1_kernel<<<128, 256, 0, stream>>>(v, x, vxm,
            leaf(0, i, 6), leaf(0, i, 7), leaf(0, i, 0), leaf(0, i, 3),
            hz1, stats + 0 * 256, i, in_dim);
        gemm1_kernel<<<128, 256, 0, stream>>>(v, x, vxm,
            leaf(1, i, 6), leaf(1, i, 7), leaf(1, i, 0), leaf(1, i, 3),
            ht1, stats + 1 * 256, i, in_dim);
        finalize_bn<<<1, 128, 0, stream>>>(stats + 0 * 256, leaf(0, i, 8),  leaf(0, i, 9),  aff + 0 * 256, 32);
        finalize_bn<<<1, 128, 0, stream>>>(stats + 1 * 256, leaf(1, i, 8),  leaf(1, i, 9),  aff + 1 * 256, 32);

        gemm32_kernel<<<128, 256, 0, stream>>>(hz1, aff + 0 * 256,
            leaf(0, i, 1), leaf(0, i, 4), hz2, stats + 2 * 256, 32, 32);
        gemm32_kernel<<<128, 256, 0, stream>>>(ht1, aff + 1 * 256,
            leaf(1, i, 1), leaf(1, i, 4), ht2, stats + 3 * 256, 32, 32);
        finalize_bn<<<1, 128, 0, stream>>>(stats + 2 * 256, leaf(0, i, 10), leaf(0, i, 11), aff + 2 * 256, 32);
        finalize_bn<<<1, 128, 0, stream>>>(stats + 3 * 256, leaf(1, i, 10), leaf(1, i, 11), aff + 3 * 256, 32);

        gemm32_kernel<<<128, 256, 0, stream>>>(hz2, aff + 2 * 256,
            leaf(0, i, 2), leaf(0, i, 5), hz3, stats + 4 * 256, outd, Cp);
        gemm32_kernel<<<128, 256, 0, stream>>>(ht2, aff + 3 * 256,
            leaf(1, i, 2), leaf(1, i, 5), ht3, stats + 5 * 256, outd, Cp);
        finalize_bn<<<1, 128, 0, stream>>>(stats + 4 * 256, leaf(0, i, 12), leaf(0, i, 13), aff + 4 * 256, outd);
        finalize_bn<<<1, 128, 0, stream>>>(stats + 5 * 256, leaf(1, i, 12), leaf(1, i, 13), aff + 5 * 256, outd);

        y_update<<<dim3(BSZ / 256, outd), 256, 0, stream>>>(y, hz3, ht3,
            aff + 4 * 256, aff + 5 * 256, w, wp, i, off, Cp);

        if ((i + 1) % 5 == 0) {
            loss_kernel<<<64, 256, 0, stream>>>(x, Kv, y, out, (i + 1) / 5);
        }
    }
}
